// Sep_Octahedral_GroupConv3d_82188494176382
// MI455X (gfx1250) — compile-verified, matrix operand order checked by reading, not run on hardware
//
#include <hip/hip_runtime.h>
#include <hip/hip_bf16.h>

// ---------------------------------------------------------------------------
// Sep_Octahedral_GroupConv3d for MI455X (gfx1250, wave32, WMMA)
//
// Shapes: B=2, I=16, O=16, G=24, H=W=D=32, K=3
//   x: (2,16,24,32,32,32) f32 (~100 MB)   out: same shape
//
// Stage 1 (WMMA GEMM): y[b,o,z,s] = sum_{g,i} g_weight[SHIFT[z,g],o,i]*x[b,i,g,s] + g_bias[o]
//   Per (b,z): M=16, K=384, N=32768 -> v_wmma_f32_16x16x32_bf16, 12 k-tiles.
//   A bank (24 z * 16x384 bf16 = 288 KB, pre-swizzled to WMMA lane layout)
//   staged into LDS with global_load_async_to_lds_b128 (ASYNCcnt path).
//   Each wave keeps its 12 B-tiles (96 VGPRs) resident and reuses them for
//   all 24 z -> x is read from HBM exactly once (fits L2 anyway: 192 MB).
// Stage 2: rotated depthwise 3x3x3 conv + bias + leaky relu (stream pass).
//
// Workspace layout (d_ws):
//   [0, 294912)                      bf16 A bank (24*12 tiles * 1 KB)
//   [294912, 294912 + 100663296)     y intermediate, f32, (b,o,z,s)
// ---------------------------------------------------------------------------

#define GRP    24
#define ICH    16
#define OCH    16
#define SPAT   32768          // 32*32*32
#define KTILES 12             // 384 / 32

typedef __attribute__((ext_vector_type(16))) __bf16        v16bf;
typedef __attribute__((ext_vector_type(8)))  float         v8f;
typedef __attribute__((ext_vector_type(8)))  unsigned int  v8u;

// ------------------------- compile-time group tables -----------------------
struct OctTables {
    int shift[24][24];   // SHIFT[g][h] = index of R_g^T R_h
    int rot[24][27];     // ROT_IDX[g][j]
};

static constexpr OctTables make_oct() {
    OctTables t{};
    int mats[24][9] = {};
    int nm = 0;
    const int perms[6][3] = {{0,1,2},{0,2,1},{1,0,2},{1,2,0},{2,0,1},{2,1,0}};
    for (int p = 0; p < 6; ++p) {
        for (int sb = 0; sb < 8; ++sb) {               // product((1,-1),repeat=3)
            const int signs[3] = { (sb & 4) ? -1 : 1, (sb & 2) ? -1 : 1, (sb & 1) ? -1 : 1 };
            int M[9] = {0,0,0,0,0,0,0,0,0};
            for (int i = 0; i < 3; ++i) M[i*3 + perms[p][i]] = signs[i];
            const int det = M[0]*(M[4]*M[8]-M[5]*M[7])
                          - M[1]*(M[3]*M[8]-M[5]*M[6])
                          + M[2]*(M[3]*M[7]-M[4]*M[6]);
            if (det == 1) { for (int j = 0; j < 9; ++j) mats[nm][j] = M[j]; ++nm; }
        }
    }
    for (int g = 0; g < 24; ++g)
        for (int h = 0; h < 24; ++h) {
            int P[9] = {};
            for (int r = 0; r < 3; ++r)
                for (int c = 0; c < 3; ++c) {
                    int s = 0;
                    for (int k = 0; k < 3; ++k) s += mats[g][k*3 + r] * mats[h][k*3 + c];
                    P[r*3 + c] = s;
                }
            for (int m = 0; m < 24; ++m) {
                bool eq = true;
                for (int j = 0; j < 9; ++j) if (mats[m][j] != P[j]) { eq = false; break; }
                if (eq) { t.shift[g][h] = m; break; }
            }
        }
    for (int g = 0; g < 24; ++g) {
        int n = 0;
        for (int X = 0; X < 3; ++X)
            for (int Y = 0; Y < 3; ++Y)
                for (int Z = 0; Z < 3; ++Z) {
                    const int v0 = X - 1, v1 = Y - 1, v2 = Z - 1;
                    const int s0 = v0*mats[g][0] + v1*mats[g][3] + v2*mats[g][6];
                    const int s1 = v0*mats[g][1] + v1*mats[g][4] + v2*mats[g][7];
                    const int s2 = v0*mats[g][2] + v1*mats[g][5] + v2*mats[g][8];
                    t.rot[g][n++] = (s0 + 1)*9 + (s1 + 1)*3 + (s2 + 1);
                }
    }
    return t;
}

__constant__ OctTables OCT = make_oct();

// f32 -> bf16 round-to-nearest-even (returned in low 16 bits)
__device__ __forceinline__ unsigned int f2bf(float f) {
    const unsigned int u = __builtin_bit_cast(unsigned int, f);
    return (u + 0x7FFFu + ((u >> 16) & 1u)) >> 16;
}

// K index (low element) inside a 16-bit 16x32 WMMA operand, per ISA layout:
// lane half selects K+8 block; dwords 0..3 -> K 0..7, dwords 4..7 -> K 16..23.
__device__ __forceinline__ int klo_of(int j, int half) {
    return ((j >= 4) ? 16 : 0) + half * 8 + ((2 * j) & 7);
}

// ---------------------- kernel 1: build bf16 A bank ------------------------
// abank[(z*12+kt)*32 + lane] = 8 dwords = 16 bf16, the exact per-lane WMMA A
// operand for A[m][k] = g_weight[SHIFT[z][k/16]][m][k%16].
__global__ void prep_abank(const float* __restrict__ gw,
                           unsigned int* __restrict__ abank) {
    const int t = blockIdx.x * blockDim.x + threadIdx.x;
    if (t >= GRP * KTILES * 32) return;
    const int lane = t & 31;
    const int kt   = (t >> 5) % KTILES;
    const int z    = t / (KTILES * 32);
    const int half = lane >> 4;
    const int m    = lane & 15;

    unsigned int outv[8];
#pragma unroll
    for (int j = 0; j < 8; ++j) {
        int k = kt * 32 + klo_of(j, half);
        int g = k >> 4, i = k & 15;
        const float flo = gw[OCT.shift[z][g] * 256 + m * 16 + i];
        ++k; g = k >> 4; i = k & 15;
        const float fhi = gw[OCT.shift[z][g] * 256 + m * 16 + i];
        outv[j] = (f2bf(fhi) << 16) | f2bf(flo);
    }
    unsigned int* dst = abank + (size_t)t * 8;
#pragma unroll
    for (int j = 0; j < 8; ++j) dst[j] = outv[j];
}

// ---------------------- kernel 2: group-mix WMMA GEMM ----------------------
// Grid: 512 blocks x 256 threads (8 waves). Wave gw -> (b, 16-wide n slice).
// Dynamic LDS: 288 KB A bank, staged via async global->LDS (no VGPR round
// trip; overlaps with the B-tile prologue loads below).
__global__ __launch_bounds__(256, 1)
void gemm_groupmix(const float* __restrict__ x,
                   const unsigned int* __restrict__ abank,
                   const float* __restrict__ gbias,
                   float* __restrict__ y) {
    extern __shared__ unsigned int lds[];   // 73728 dwords = 288 KB

    // ---- async A-bank stage: 18432 x b128 over 256 threads (72 each) ----
    {
        const int tid = threadIdx.x;
        for (int i = tid; i < GRP * KTILES * 32 * 2; i += 256) {
            const unsigned long long gaddr =
                (unsigned long long)(const void*)(abank + (size_t)i * 4);
            const unsigned int laddr =
                (unsigned int)(unsigned long long)(const void*)(lds + (size_t)i * 4);
            asm volatile("global_load_async_to_lds_b128 %0, %1, off"
                         :: "v"(laddr), "v"(gaddr) : "memory");
        }
    }

    const int wave = threadIdx.x >> 5;
    const int lane = threadIdx.x & 31;
    const int half = lane >> 4;
    const int n    = lane & 15;

    const int gw = blockIdx.x * 8 + wave;        // 4096 waves total
    const int b  = gw >> 11;                     // 2048 slices per batch
    const int s0 = (gw & 2047) << 4;

    const float* xb = x + (size_t)b * ICH * GRP * SPAT + (s0 + n);

    // Load all 12 B tiles once (f32 -> bf16), keep resident: reused for 24 z.
    // These global loads overlap with the async LDS fill above.
    v8u bt[KTILES];
#pragma unroll
    for (int kt = 0; kt < KTILES; ++kt) {
#pragma unroll
        for (int j = 0; j < 8; ++j) {
            int k = kt * 32 + klo_of(j, half);
            const float flo = xb[((size_t)(k & 15) * GRP + (k >> 4)) * SPAT];
            ++k;
            const float fhi = xb[((size_t)(k & 15) * GRP + (k >> 4)) * SPAT];
            bt[kt][j] = (f2bf(fhi) << 16) | f2bf(flo);
        }
    }

    float bias[8];
#pragma unroll
    for (int r = 0; r < 8; ++r) bias[r] = gbias[r + 8 * half];

    // async fill must be complete (and visible workgroup-wide) before ds reads
    asm volatile("s_wait_asynccnt 0x0" ::: "memory");
    __syncthreads();

    const v8u* Ab = (const v8u*)lds;
    for (int z = 0; z < GRP; ++z) {
        v8f acc = {};
#pragma unroll
        for (int kt = 0; kt < KTILES; ++kt) {
            const v8u a = Ab[(z * KTILES + kt) * 32 + lane];
            acc = __builtin_amdgcn_wmma_f32_16x16x32_bf16(
                      false, __builtin_bit_cast(v16bf, a),
                      false, __builtin_bit_cast(v16bf, bt[kt]),
                      (short)0, acc, false, false);
        }
        // C/D layout: vgpr r, lanes 0-15 -> o=r, lanes 16-31 -> o=r+8
        float* yb = y + ((size_t)b * (OCH * GRP) + z) * SPAT + (s0 + n);
#pragma unroll
        for (int r = 0; r < 8; ++r)
            yb[(size_t)(r + 8 * half) * GRP * SPAT] = acc[r] + bias[r];
    }
}

// ------------- kernel 3: rotated depthwise 3x3x3 + bias + lrelu ------------
// One block = 256 outputs of a fixed channel c=(b,o,z); 32768 % 256 == 0 so
// all threads in a block share (o,z) -> stage the 27 rotated taps in LDS.
__global__ __launch_bounds__(256)
void dwconv_lrelu(const float* __restrict__ y,
                  const float* __restrict__ tw,     // (16,27)
                  const float* __restrict__ tbias,
                  float* __restrict__ out) {
    __shared__ float tk[27];

    const int c      = blockIdx.x >> 7;                       // (b*16+o)*24+z
    const int within = ((blockIdx.x & 127) << 8) + threadIdx.x;
    const int z = c % GRP;
    const int o = (c / GRP) & 15;

    if (threadIdx.x < 27)
        tk[threadIdx.x] = tw[o * 27 + OCT.rot[z][threadIdx.x]];
    __syncthreads();

    const int d = within & 31;
    const int w = (within >> 5) & 31;
    const int h = within >> 10;

    const float* yc = y + (size_t)c * SPAT;
    float sum = 0.0f;
    int j = 0;
#pragma unroll
    for (int dh = -1; dh <= 1; ++dh)
#pragma unroll
        for (int dw = -1; dw <= 1; ++dw)
#pragma unroll
            for (int dd = -1; dd <= 1; ++dd, ++j) {
                const int hh = h + dh, ww = w + dw, zz = d + dd;
                if ((unsigned)hh < 32u && (unsigned)ww < 32u && (unsigned)zz < 32u)
                    sum += tk[j] * yc[(hh << 10) + (ww << 5) + zz];
            }
    const float v = sum + tbias[0];
    out[(size_t)c * SPAT + within] = (v > 0.0f) ? v : 0.01f * v;
}

// ---------------------------------------------------------------------------
extern "C" void kernel_launch(void* const* d_in, const int* in_sizes, int n_in,
                              void* d_out, int out_size, void* d_ws, size_t ws_size,
                              hipStream_t stream) {
    const float* x  = (const float*)d_in[0];   // (2,16,24,32,32,32)
    const float* gw = (const float*)d_in[1];   // (24,16,16)
    const float* tw = (const float*)d_in[2];   // (16,27)
    const float* gb = (const float*)d_in[3];   // (16,)
    const float* tb = (const float*)d_in[4];   // (1,)
    float* out = (float*)d_out;

    unsigned int* abank = (unsigned int*)d_ws;                       // 294912 B
    float*        y     = (float*)((char*)d_ws + 294912);            // 100663296 B

    // 1) A-bank prep: 24*12*32 = 9216 threads
    prep_abank<<<36, 256, 0, stream>>>(gw, abank);

    // 2) WMMA GEMM: 4096 waves = 512 blocks x 8 waves, 288 KB LDS each
    const size_t lds_bytes = (size_t)GRP * KTILES * 32 * 32;         // 294912
    gemm_groupmix<<<512, 256, lds_bytes, stream>>>(x, abank, gb, y);

    // 3) depthwise conv + bias + leaky relu: 768 channels * 128 blocks
    dwconv_lrelu<<<768 * 128, 256, 0, stream>>>(y, tw, tb, out);
}
